// GPTQLinear_39213051412616
// MI455X (gfx1250) — compile-verified, matrix operand order checked by reading, not run on hardware
//
#include <hip/hip_runtime.h>

// GPTQ 4-bit dequant + fp16 WMMA GEMM for gfx1250 (MI455X).
// M=8192 (4*2048), K=4096, N=4096, group size 128.
// K-step 64 (two 16x16x32 k-subtiles per barrier), magic-number fp16 dequant,
// register-pipelined global prefetch under the WMMA section.

#define K_DIM 4096
#define N_DIM 4096
#define M_DIM 8192
#define KSTEP 64
#define NIT   (K_DIM / KSTEP)

typedef __attribute__((ext_vector_type(16))) _Float16 v16h;
typedef __attribute__((ext_vector_type(2)))  _Float16 h2;
typedef __attribute__((ext_vector_type(8)))  float    v8f;

__global__ __launch_bounds__(256) void gptq_f16_wmma(
    const float* __restrict__ x,      // [M, K] fp32
    const int*   __restrict__ qw,     // [K/8, N] packed nibbles along K
    const int*   __restrict__ qz,     // [K/128, N/8] packed nibbles along N
    const float* __restrict__ sc,     // [K/128, N]
    const float* __restrict__ bias,   // [N]
    float*       __restrict__ out)    // [M, N] fp32
{
  // Fragment-order LDS: block = 1024B, addr = block*1024 + lane*32 + e*2.
  // A: mfrag(0..7) x ks(0..1) blocks in [0,16K); B: nfrag x ks in [16K,32K).
  __shared__ __align__(32) unsigned char lds[32768];

  const int tid  = threadIdx.x;
  const int lane = tid & 31;
  const int wave = tid >> 5;
  const int wm   = wave >> 1;          // 0..3 : 32-row M sub-block
  const int wn   = wave &  1;          // 0..1 : 64-col N sub-block
  const int n0   = blockIdx.x * 128;
  const int m0   = blockIdx.y * 128;

  v8f acc[2][4];
  const v8f vzero = {0.f, 0.f, 0.f, 0.f, 0.f, 0.f, 0.f, 0.f};
#pragma unroll
  for (int mi = 0; mi < 2; ++mi)
#pragma unroll
    for (int ni = 0; ni < 4; ++ni) acc[mi][ni] = vzero;

  // ---- A writer: thread (row=tid>>1, half=tid&1) ----
  // Within a k-subtile, half h supplies K = {8h..8h+7}(e=0..7) U {16+8h..+7}(e=8..15),
  // landing in lane = 16h + row%16 => two contiguous 16B stores per subtile.
  const int ar = tid >> 1;
  const int ah = tid & 1;
  const float* xrow = x + (size_t)(m0 + ar) * K_DIM;
  unsigned char* a_dst = lds + ((ar >> 4) * 2) * 1024 + ((ah << 4) + (ar & 15)) * 32;

  // ---- B writer: thread handles qweight rows kt*8 + kbl + 2h (h=0..3) of one col.
  // Word -> lane = 16*kbl + oc%16, subtile ks = h>>1, e-base = 8*(h&1)
  // => one contiguous 16B store per word.
  const int oc     = tid & 127;
  const int kbl    = tid >> 7;                  // 0 or 1
  const int ocol   = n0 + oc;
  const int zshift = (ocol & 7) * 4;
  unsigned char* b_dst = lds + 16384 + ((oc >> 4) * 2) * 1024
                       + ((kbl << 4) + (oc & 15)) * 32;

  // ---- prefetch registers ----
  float4   pf[8];     // 32 floats of x
  unsigned pwv[4];    // 4 packed qweight words
  float    ps, pzf;   // scale, zero+1 (float)

  auto loadA = [&](int kt) {
    const float* p = xrow + kt * KSTEP + 8 * ah;
#pragma unroll
    for (int ks = 0; ks < 2; ++ks)
#pragma unroll
      for (int blk = 0; blk < 2; ++blk) {
        const float4* q = (const float4*)(p + ks * 32 + blk * 16);
        pf[(ks * 2 + blk) * 2 + 0] = q[0];
        pf[(ks * 2 + blk) * 2 + 1] = q[1];
      }
  };
  auto loadB = [&](int kt) {
    const int g = kt >> 1;                      // 64 | 128 => group constant per kt
    ps  = sc[(size_t)g * N_DIM + ocol];
    pzf = (float)(((((unsigned)qz[(size_t)g * (N_DIM / 8) + (ocol >> 3)])
                     >> zshift) & 15u) + 1u);
#pragma unroll
    for (int h = 0; h < 4; ++h)
      pwv[h] = (unsigned)qw[(size_t)(kt * 8 + kbl + 2 * h) * N_DIM + ocol];
  };

  auto writeA = [&]() {
#pragma unroll
    for (int ks = 0; ks < 2; ++ks) {
      const float4 f0 = pf[ks * 4 + 0], f1 = pf[ks * 4 + 1];
      const float4 f2 = pf[ks * 4 + 2], f3 = pf[ks * 4 + 3];
      int4 v0, v1;
      v0.x = __builtin_bit_cast(int, __builtin_amdgcn_cvt_pkrtz(f0.x, f0.y));
      v0.y = __builtin_bit_cast(int, __builtin_amdgcn_cvt_pkrtz(f0.z, f0.w));
      v0.z = __builtin_bit_cast(int, __builtin_amdgcn_cvt_pkrtz(f1.x, f1.y));
      v0.w = __builtin_bit_cast(int, __builtin_amdgcn_cvt_pkrtz(f1.z, f1.w));
      v1.x = __builtin_bit_cast(int, __builtin_amdgcn_cvt_pkrtz(f2.x, f2.y));
      v1.y = __builtin_bit_cast(int, __builtin_amdgcn_cvt_pkrtz(f2.z, f2.w));
      v1.z = __builtin_bit_cast(int, __builtin_amdgcn_cvt_pkrtz(f3.x, f3.y));
      v1.w = __builtin_bit_cast(int, __builtin_amdgcn_cvt_pkrtz(f3.z, f3.w));
      *(int4*)(a_dst + ks * 1024)      = v0;   // e = 0..7
      *(int4*)(a_dst + ks * 1024 + 16) = v1;   // e = 8..15
    }
  };
  auto writeB = [&]() {
    // magic: 0x6400|nib == (fp16)(1024+nib); dequant = s*((1024+nib)-(1024+z))
    const _Float16 hs = (_Float16)ps;
    const _Float16 hz = (_Float16)(1024.0f + pzf);
    const h2 hs2 = {hs, hs};
    const h2 hz2 = {hz, hz};
#pragma unroll
    for (int h = 0; h < 4; ++h) {
      const unsigned w = pwv[h];
      int4 bv;
#pragma unroll
      for (int p = 0; p < 4; ++p) {
        const unsigned pw = w >> (8 * p);
        const unsigned hb = (pw & 0xFu) | ((pw & 0xF0u) << 12) | 0x64006400u;
        const h2 hv = __builtin_bit_cast(h2, hb);
        const h2 r  = (hv - hz2) * hs2;
        ((unsigned*)&bv)[p] = __builtin_bit_cast(unsigned, r);
      }
      *(int4*)(b_dst + (h >> 1) * 1024 + (h & 1) * 16) = bv;
    }
  };

  loadA(0);
  loadB(0);

  for (int kt = 0; kt < NIT; ++kt) {
    writeA();
    writeB();
    __syncthreads();

    // Prefetch next tile's globals; latency hides under the WMMA section.
    if (kt + 1 < NIT) { loadA(kt + 1); loadB(kt + 1); }

#pragma unroll
    for (int ks = 0; ks < 2; ++ks) {
      const v16h a0 = *(const v16h*)(lds + (((wm * 2 + 0) * 2 + ks)) * 1024 + lane * 32);
      const v16h a1 = *(const v16h*)(lds + (((wm * 2 + 1) * 2 + ks)) * 1024 + lane * 32);
      const v16h b0 = *(const v16h*)(lds + 16384 + (((wn * 4 + 0) * 2 + ks)) * 1024 + lane * 32);
      const v16h b1 = *(const v16h*)(lds + 16384 + (((wn * 4 + 1) * 2 + ks)) * 1024 + lane * 32);
      const v16h b2 = *(const v16h*)(lds + 16384 + (((wn * 4 + 2) * 2 + ks)) * 1024 + lane * 32);
      const v16h b3 = *(const v16h*)(lds + 16384 + (((wn * 4 + 3) * 2 + ks)) * 1024 + lane * 32);
      acc[0][0] = __builtin_amdgcn_wmma_f32_16x16x32_f16(false, a0, false, b0, (short)0, acc[0][0], false, false);
      acc[1][0] = __builtin_amdgcn_wmma_f32_16x16x32_f16(false, a1, false, b0, (short)0, acc[1][0], false, false);
      acc[0][1] = __builtin_amdgcn_wmma_f32_16x16x32_f16(false, a0, false, b1, (short)0, acc[0][1], false, false);
      acc[1][1] = __builtin_amdgcn_wmma_f32_16x16x32_f16(false, a1, false, b1, (short)0, acc[1][1], false, false);
      acc[0][2] = __builtin_amdgcn_wmma_f32_16x16x32_f16(false, a0, false, b2, (short)0, acc[0][2], false, false);
      acc[1][2] = __builtin_amdgcn_wmma_f32_16x16x32_f16(false, a1, false, b2, (short)0, acc[1][2], false, false);
      acc[0][3] = __builtin_amdgcn_wmma_f32_16x16x32_f16(false, a0, false, b3, (short)0, acc[0][3], false, false);
      acc[1][3] = __builtin_amdgcn_wmma_f32_16x16x32_f16(false, a1, false, b3, (short)0, acc[1][3], false, false);
    }

    __syncthreads();
  }

  // ---- epilogue: C layout (lane%16 -> N, vgpr v + 8*(lane>=16) -> M) ----
  const int half = lane >> 4;
  const int cn   = lane & 15;
#pragma unroll
  for (int ni = 0; ni < 4; ++ni) {
    const int   col = n0 + wn * 64 + ni * 16 + cn;
    const float bv  = bias[col];
#pragma unroll
    for (int mi = 0; mi < 2; ++mi) {
      const int rbase = m0 + wm * 32 + mi * 16 + half * 8;
#pragma unroll
      for (int v = 0; v < 8; ++v) {
        out[(size_t)(rbase + v) * N_DIM + col] = acc[mi][ni][v] + bv;
      }
    }
  }
}

extern "C" void kernel_launch(void* const* d_in, const int* in_sizes, int n_in,
                              void* d_out, int out_size, void* d_ws, size_t ws_size,
                              hipStream_t stream) {
  const float* x    = (const float*)d_in[0];
  const int*   qw   = (const int*)  d_in[1];
  const int*   qz   = (const int*)  d_in[2];
  const float* sc   = (const float*)d_in[3];
  const float* bias = (const float*)d_in[4];
  float*       out  = (float*)d_out;

  dim3 grid(N_DIM / 128, M_DIM / 128);   // 32 x 64 tiles
  gptq_f16_wmma<<<grid, 256, 0, stream>>>(x, qw, qz, sc, bias, out);
}
